// CQAttention_84851373900367
// MI455X (gfx1250) — compile-verified, matrix operand order checked by reading, not run on hardware
//
#include <hip/hip_runtime.h>

// ---------------------------------------------------------------------------
// CQAttention for MI455X (gfx1250, wave32, WMMA f32_16x16x32_f16)
// B=64, D=128, LC=1000 (pad 1008/1024), LQ=100 (pad 112/128)
// ---------------------------------------------------------------------------

typedef __attribute__((ext_vector_type(16))) _Float16 v16h;
typedef __attribute__((ext_vector_type(8)))  _Float16 v8h;
typedef __attribute__((ext_vector_type(8)))  float    v8f;

union V16H { v16h v; v8h h[2]; };
union V8F  { v8f v; float e[8]; };
union V8H  { v8h v; _Float16 e[8]; };

#define NB   64
#define DD   128
#define LC   1000
#define LQ   100
#define LQP  112      // 7 * 16
#define NIT  63       // ceil(1000/16)
#define NJT  7        // ceil(100/16)

// Workspace layout (bytes, all offsets 256-aligned)
#define WS_S     ((size_t)0)            // f32 [64][1000][112]   = 28,672,000 B
#define WS_RMAX  ((size_t)28672000)     // f32 [64000]           =    256,000 B
#define WS_RSUM  ((size_t)28928000)     // f32 [64000]           =    256,000 B
#define WS_CMAX  ((size_t)29184000)     // f32 [64][112]         =     28,672 B
#define WS_CSUM  ((size_t)29212672)     // f32 [64][112]         =     28,672 B
#define WS_T     ((size_t)29241344)     // f16 [64][128][128]    =  2,097,152 B
#define WS_QH    ((size_t)33435648)     // f16 [64][112][128]    =  1,835,008 B
#define WS_QG    ((size_t)35270656)     // f16 [64][128][128]    =  2,097,152 B
#define WS_CH    ((size_t)39464960)     // f16 [64][128][1024]   = 16,777,216 B
// total = 56,242,176 B

// Per-ISA 16-bit A/B fragment layout: lane L holds row (L&15); halves 0..7 are
// K = kb..kb+7 and halves 8..15 are K = kb+16..kb+23 where kb = k0 + 8*(L>=16).
// Two contiguous 16-byte loads per fragment.
__device__ __forceinline__ v16h frag_load(const _Float16* row, int k0, int lane) {
  V16H r;
  int kb = k0 + ((lane >> 4) << 3);
  r.h[0] = *(const v8h*)(row + kb);
  r.h[1] = *(const v8h*)(row + kb + 16);
  return r.v;
}

__device__ __forceinline__ v8f wmma16(v16h a, v16h b, v8f c) {
  return __builtin_amdgcn_wmma_f32_16x16x32_f16(false, a, false, b, (short)0, c,
                                                false, false);
}

// ---------------------------------------------------------------------------
// Kernel 0: precompute f16 padded operand layouts; zero T.
// ---------------------------------------------------------------------------
__global__ void cqa_prep(const float* __restrict__ C, const float* __restrict__ Q,
                         _Float16* __restrict__ Qh, _Float16* __restrict__ Qg,
                         _Float16* __restrict__ Ch, _Float16* __restrict__ T) {
  const int stride = gridDim.x * blockDim.x;
  int tid = blockIdx.x * blockDim.x + threadIdx.x;

  // Qh[b][j][d] = Q[b][d][j]  (j padded to 112)
  for (int idx = tid; idx < NB * LQP * DD; idx += stride) {
    int b = idx / (LQP * DD), r = idx % (LQP * DD);
    int j = r / DD, d = r % DD;
    float v = (j < LQ) ? Q[((size_t)(b * DD + d)) * LQ + j] : 0.f;
    Qh[idx] = (_Float16)v;
  }
  // Qg[b][d][j] = Q[b][d][j]  (j padded to 128)
  for (int idx = tid; idx < NB * DD * 128; idx += stride) {
    int b = idx >> 14, r = idx & 16383;
    int d = r >> 7, j = r & 127;
    float v = (j < LQ) ? Q[((size_t)(b * DD + d)) * LQ + j] : 0.f;
    Qg[idx] = (_Float16)v;
  }
  // Ch[b][d][i] = C[b][d][i]  (i padded to 1024)
  for (int idx = tid; idx < NB * DD * 1024; idx += stride) {
    int b = idx >> 17, r = idx & 131071;
    int d = r >> 10, i = r & 1023;
    float v = (i < LC) ? C[((size_t)(b * DD + d)) * LC + i] : 0.f;
    Ch[idx] = (_Float16)v;
  }
  // zero T (j in [112,128) stays zero; rest overwritten by cqa_t)
  for (int idx = tid; idx < NB * DD * 128; idx += stride) T[idx] = (_Float16)0.f;
}

// ---------------------------------------------------------------------------
// Kernel 1: S[b,i,j] = sum_d (w1+w3*Ct)[i,d] * Qt[j,d]  + rowbias[i]
// block = (b, i-tile of 16); 7 waves, one 16x16 j-tile each; K=128 (4 WMMAs).
// ---------------------------------------------------------------------------
__global__ void cqa_s(const float* __restrict__ C, const float* __restrict__ W,
                      const _Float16* __restrict__ Qh, float* __restrict__ S) {
  const int b = blockIdx.x / NIT;
  const int i0 = (blockIdx.x % NIT) * 16;
  const int tid = threadIdx.x;
  const int wv = tid >> 5, lane = tid & 31;

  __shared__ __attribute__((aligned(16))) _Float16 u[16][128];  // w1 + w3*Ct
  __shared__ float w2c[16][128];
  __shared__ float rs[16];

  for (int idx = tid; idx < 16 * 128; idx += 224) {
    int i = idx >> 7, d = idx & 127;
    int gi = i0 + i;
    float w1 = 0.f, w2 = 0.f, w3 = 0.f, c = 0.f;
    if (gi < LC) {
      const float* wrow = W + ((size_t)(b * LC + gi)) * 384;
      w1 = wrow[d]; w2 = wrow[128 + d]; w3 = wrow[256 + d];
      c = C[((size_t)(b * DD + d)) * LC + gi];
    }
    u[i][d] = (_Float16)(w1 + w3 * c);
    w2c[i][d] = w2 * c;
  }
  __syncthreads();
  if (tid < 16) {
    float s = 0.f;
    for (int d = 0; d < 128; ++d) s += w2c[tid][d];
    rs[tid] = s;
  }
  __syncthreads();

  const int j0 = wv * 16;
  const _Float16* arow = &u[lane & 15][0];
  const _Float16* brow = Qh + ((size_t)(b * LQP + j0 + (lane & 15))) * DD;
  v8f acc = {};
  for (int k0 = 0; k0 < 128; k0 += 32) {
    v16h a = frag_load(arow, k0, lane);
    v16h bq = frag_load(brow, k0, lane);
    acc = wmma16(a, bq, acc);
  }
  // D layout: n = lane&15, m = r + 8*(lane>=16)
  V8F av; av.v = acc;
  int j = j0 + (lane & 15);
  if (j < LQ) {
    int mb = (lane >> 4) << 3;
    for (int r = 0; r < 8; ++r) {
      int gi = i0 + mb + r;
      if (gi < LC)
        S[((size_t)(b * LC + gi)) * LQP + j] = av.e[r] + rs[mb + r];
    }
  }
}

// ---------------------------------------------------------------------------
// Kernel 2a: per-row (axis=2) softmax stats; one wave per row.
// ---------------------------------------------------------------------------
__global__ void cqa_rowstat(const float* __restrict__ S, float* __restrict__ rmax,
                            float* __restrict__ rsum) {
  int row = blockIdx.x * 8 + (threadIdx.x >> 5);   // 0 .. 63999
  int lane = threadIdx.x & 31;
  const float* p = S + (size_t)row * LQP;
  float v[4], m = -1e30f;
  for (int t = 0; t < 4; ++t) {
    int j = lane + 32 * t;
    v[t] = (j < LQ) ? p[j] : -1e30f;
    m = fmaxf(m, v[t]);
  }
  for (int off = 16; off; off >>= 1) m = fmaxf(m, __shfl_xor(m, off, 32));
  float s = 0.f;
  for (int t = 0; t < 4; ++t) {
    int j = lane + 32 * t;
    if (j < LQ) s += __expf(v[t] - m);
  }
  for (int off = 16; off; off >>= 1) s += __shfl_xor(s, off, 32);
  if (lane == 0) { rmax[row] = m; rsum[row] = s; }
}

// ---------------------------------------------------------------------------
// Kernel 2b: per-column (axis=1) softmax stats; thread per column, coalesced i.
// ---------------------------------------------------------------------------
__global__ void cqa_colstat(const float* __restrict__ S, float* __restrict__ cmax,
                            float* __restrict__ csum) {
  int b = blockIdx.x, j = threadIdx.x;
  if (j >= LQP) return;
  float m = -1e30f, s = 0.f;
  if (j < LQ) {
    const float* p = S + (size_t)b * LC * LQP + j;
    for (int i = 0; i < LC; ++i) {
      float x = p[(size_t)i * LQP];
      if (x > m) { s *= __expf(m - x); m = x; }
      s += __expf(x - m);
    }
  } else { m = 0.f; s = 1.f; }
  cmax[b * LQP + j] = m;
  csum[b * LQP + j] = s;
}

// ---------------------------------------------------------------------------
// Kernel 3: T[b][d][j] = sum_i S2[b,i,j] * Ct[b,i,d]   (M=j, N=d, K=i)
// block = (b, j-tile); 8 waves each own 16-wide d-tile; 32 K-chunks of 32.
// ---------------------------------------------------------------------------
__global__ void cqa_t(const float* __restrict__ S, const float* __restrict__ cmax,
                      const float* __restrict__ csum, const _Float16* __restrict__ Ch,
                      _Float16* __restrict__ T) {
  const int b = blockIdx.x / NJT;
  const int j0 = (blockIdx.x % NJT) * 16;
  const int tid = threadIdx.x, wv = tid >> 5, lane = tid & 31;

  __shared__ __attribute__((aligned(16))) _Float16 As[16][32];
  __shared__ float cm[16], csr[16];
  if (tid < 16) {
    cm[tid] = cmax[b * LQP + j0 + tid];
    csr[tid] = 1.0f / csum[b * LQP + j0 + tid];
  }

  const _Float16* arow = &As[lane & 15][0];
  const _Float16* brow = Ch + ((size_t)(b * DD + wv * 16 + (lane & 15))) * 1024;
  v8f acc = {};
  for (int i0 = 0; i0 < 1024; i0 += 32) {
    __syncthreads();
    for (int idx = tid; idx < 16 * 32; idx += 256) {
      int jl = idx >> 5, il = idx & 31;
      int gi = i0 + il, gj = j0 + jl;
      float e = 0.f;
      if (gi < LC && gj < LQ)
        e = __expf(S[((size_t)(b * LC + gi)) * LQP + gj] - cm[jl]);
      As[jl][il] = (_Float16)e;
    }
    __syncthreads();
    v16h a = frag_load(arow, 0, lane);
    v16h bc = frag_load(brow, i0, lane);
    acc = wmma16(a, bc, acc);
  }

  // D: n = lane&15 -> d_local, m = r + 8*(lane>=16) -> j_local (consecutive j)
  V8F av; av.v = acc;
  int d = wv * 16 + (lane & 15);
  int mb = (lane >> 4) << 3;
  V8H o;
  for (int r = 0; r < 8; ++r) {
    int jl = mb + r;
    float val = (j0 + jl < LQ) ? av.e[r] * csr[jl] : 0.f;
    o.e[r] = (_Float16)val;
  }
  *(v8h*)(T + ((size_t)(b * DD + d)) * 128 + j0 + mb) = o.v;
}

// ---------------------------------------------------------------------------
// Kernel 4: A = S1@Qt, Bm = S1@T; emit [Ct, A, Ct*A, Ct*Bm] transposed.
// block = (b, i-tile); 8 waves each own 16-wide d-tile; shared A-fragment.
// ---------------------------------------------------------------------------
__global__ void cqa_out(const float* __restrict__ C, const float* __restrict__ S,
                        const float* __restrict__ rmax, const float* __restrict__ rsum,
                        const _Float16* __restrict__ Qg, const _Float16* __restrict__ T,
                        float* __restrict__ out) {
  const int b = blockIdx.x / NIT;
  const int i0 = (blockIdx.x % NIT) * 16;
  const int tid = threadIdx.x, wv = tid >> 5, lane = tid & 31;

  __shared__ __attribute__((aligned(16))) _Float16 As1[16][128];
  __shared__ float A_s[16][128];
  __shared__ float Bm_s[16][128];

  for (int idx = tid; idx < 16 * 128; idx += 256) {
    int m = idx >> 7, j = idx & 127;
    int gi = i0 + m;
    float e = 0.f;
    if (gi < LC && j < LQ) {
      int row = b * LC + gi;
      e = __expf(S[(size_t)row * LQP + j] - rmax[row]) / rsum[row];
    }
    As1[m][j] = (_Float16)e;
  }
  __syncthreads();

  const int d = wv * 16 + (lane & 15);
  const _Float16* arow = &As1[lane & 15][0];
  const _Float16* browQ = Qg + ((size_t)(b * DD + d)) * 128;
  const _Float16* browT = T + ((size_t)(b * DD + d)) * 128;
  v8f accA = {}, accB = {};
  for (int k0 = 0; k0 < 128; k0 += 32) {
    v16h a = frag_load(arow, k0, lane);
    v16h bq = frag_load(browQ, k0, lane);
    v16h bt = frag_load(browT, k0, lane);
    accA = wmma16(a, bq, accA);
    accB = wmma16(a, bt, accB);
  }
  V8F ua, ub; ua.v = accA; ub.v = accB;
  int mb = (lane >> 4) << 3;
  for (int r = 0; r < 8; ++r) {
    A_s[mb + r][d] = ua.e[r];
    Bm_s[mb + r][d] = ub.e[r];
  }
  __syncthreads();

  for (int idx = tid; idx < 512 * 16; idx += 256) {
    int c = idx >> 4, il = idx & 15;
    int gi = i0 + il;
    if (gi >= LC) continue;
    int dd = c & 127, sec = c >> 7;
    float ct = C[((size_t)(b * DD + dd)) * LC + gi];
    float val;
    if (sec == 0)      val = ct;
    else if (sec == 1) val = A_s[il][dd];
    else if (sec == 2) val = ct * A_s[il][dd];
    else               val = ct * Bm_s[il][dd];
    out[((size_t)(b * 512 + c)) * LC + gi] = val;
  }
}

// ---------------------------------------------------------------------------
extern "C" void kernel_launch(void* const* d_in, const int* in_sizes, int n_in,
                              void* d_out, int out_size, void* d_ws, size_t ws_size,
                              hipStream_t stream) {
  (void)in_sizes; (void)n_in; (void)out_size; (void)ws_size;
  const float* C = (const float*)d_in[0];
  const float* Q = (const float*)d_in[1];
  const float* W = (const float*)d_in[2];
  float* out = (float*)d_out;
  char* ws = (char*)d_ws;

  float*    Sbuf = (float*)(ws + WS_S);
  float*    rmax = (float*)(ws + WS_RMAX);
  float*    rsum = (float*)(ws + WS_RSUM);
  float*    cmax = (float*)(ws + WS_CMAX);
  float*    csum = (float*)(ws + WS_CSUM);
  _Float16* T    = (_Float16*)(ws + WS_T);
  _Float16* Qh   = (_Float16*)(ws + WS_QH);
  _Float16* Qg   = (_Float16*)(ws + WS_QG);
  _Float16* Ch   = (_Float16*)(ws + WS_CH);

  cqa_prep<<<1024, 256, 0, stream>>>(C, Q, Qh, Qg, Ch, T);
  cqa_s<<<NB * NIT, 224, 0, stream>>>(C, W, Qh, Sbuf);
  cqa_rowstat<<<(NB * LC) / 8, 256, 0, stream>>>(Sbuf, rmax, rsum);
  cqa_colstat<<<NB, 128, 0, stream>>>(Sbuf, cmax, csum);
  cqa_t<<<NB * NJT, 256, 0, stream>>>(Sbuf, cmax, csum, Ch, T);
  cqa_out<<<NB * NIT, 256, 0, stream>>>(C, Sbuf, rmax, rsum, Qg, T, out);
}
